// JointEncoder_33165737459943
// MI455X (gfx1250) — matrix operand order repaired and checked by reading, not compiled, not run on hardware
//
#include <hip/hip_runtime.h>

typedef __attribute__((ext_vector_type(16))) _Float16 v16h;
typedef __attribute__((ext_vector_type(8)))  _Float16 v8h;
typedef __attribute__((ext_vector_type(8)))  float    v8f;

#define PN2_B 8
#define PN2_NPER 2048
#define PN2_N1 2047
#define PN2_N2 676
#define PN2_K 64

// ---------------------------------------------------------------------------
// Weight conversion: W [Cin,Nout] f32 row-major -> Wt [Nout,Kpad] f16 (transposed,
// zero padded along K). Makes WMMA B-operand loads contiguous per lane.
// ---------------------------------------------------------------------------
__global__ void pn2_wt_convert(const float* __restrict__ W, _Float16* __restrict__ Wt,
                               int Cin, int Nout, int Kpad) {
    int t = blockIdx.x * blockDim.x + threadIdx.x;
    int total = Nout * Kpad;
    if (t >= total) return;
    int n = t / Kpad, k = t % Kpad;
    Wt[t] = (k < Cin) ? (_Float16)W[(size_t)k * Nout + n] : (_Float16)0.f;
}

// ---------------------------------------------------------------------------
// Farthest point sampling. One workgroup (256 threads) per batch; points + dmin
// live in LDS (CDNA5: 320KB/WGP, we use ~35KB). 'nsamp' sequential argmax steps,
// tie-break lowest index to match jnp.argmax.
// ---------------------------------------------------------------------------
__global__ void pn2_fps(const float* __restrict__ pos, int* __restrict__ out_idx,
                        int Nsrc, int nsamp) {
    __shared__ float px[2048], py[2048], pz[2048], dmin[2048];
    __shared__ float rv[256];
    __shared__ int   ri[256];
    __shared__ int   s_last;
    int b   = blockIdx.x;
    int tid = threadIdx.x;
    const float* p = pos + (size_t)b * Nsrc * 3;
    for (int i = tid; i < Nsrc; i += 256) {
        px[i] = p[i * 3 + 0]; py[i] = p[i * 3 + 1]; pz[i] = p[i * 3 + 2];
        dmin[i] = 1e10f;
    }
    __syncthreads();
    int last = 0;
    for (int s = 0; s < nsamp; ++s) {
        if (tid == 0) out_idx[(size_t)b * nsamp + s] = last;
        float lx = px[last], ly = py[last], lz = pz[last];
        float bv = -1.f; int bi = 0x7fffffff;
        for (int i = tid; i < Nsrc; i += 256) {
            float dx = px[i] - lx, dy = py[i] - ly, dz = pz[i] - lz;
            float d = dx * dx + dy * dy + dz * dz;
            float dm = dmin[i];
            dm = fminf(dm, d);
            dmin[i] = dm;
            if (dm > bv || (dm == bv && i < bi)) { bv = dm; bi = i; }
        }
        rv[tid] = bv; ri[tid] = bi;
        __syncthreads();
        for (int off = 128; off > 0; off >>= 1) {
            if (tid < off) {
                float ov = rv[tid + off]; int oi = ri[tid + off];
                if (ov > rv[tid] || (ov == rv[tid] && oi < ri[tid])) { rv[tid] = ov; ri[tid] = oi; }
            }
            __syncthreads();
        }
        if (tid == 0) s_last = ri[0];
        __syncthreads();
        last = s_last;
        __syncthreads();
    }
}

// gather sampled positions: dst[b*M+m] = src[b*Nsrc + idx[b*M+m]]
__global__ void pn2_gather3(const float* __restrict__ src, const int* __restrict__ idx,
                            float* __restrict__ dst, int M, int Nsrc, int total) {
    int t = blockIdx.x * blockDim.x + threadIdx.x;
    if (t >= total) return;
    int b = t / M;
    int j = idx[t];
    const float* s = src + ((size_t)b * Nsrc + j) * 3;
    dst[t * 3 + 0] = s[0]; dst[t * 3 + 1] = s[1]; dst[t * 3 + 2] = s[2];
}

// ---------------------------------------------------------------------------
// Ball query (k smallest d2, radius mask). One wave32 per query; d2 in LDS;
// 64 iterative wave min-with-index reductions (__shfl_xor), tie lowest index.
// ---------------------------------------------------------------------------
__global__ void pn2_ball_query(const float* __restrict__ src, const float* __restrict__ qry,
                               int Nsrc, int Mq, int totalQ, float r2,
                               int* __restrict__ nidx, unsigned char* __restrict__ mask) {
    __shared__ float d2s[2048];
    int q = blockIdx.x;
    if (q >= totalQ) return;
    int lane = threadIdx.x;
    int b = q / Mq;
    const float* sp = src + (size_t)b * Nsrc * 3;
    float qx = qry[q * 3 + 0], qy = qry[q * 3 + 1], qz = qry[q * 3 + 2];
    for (int i = lane; i < 2048; i += 32) {
        float d = 3.0e38f;
        if (i < Nsrc) {
            float dx = sp[i * 3 + 0] - qx, dy = sp[i * 3 + 1] - qy, dz = sp[i * 3 + 2] - qz;
            d = dx * dx + dy * dy + dz * dz;
        }
        d2s[i] = d;
    }
    __syncthreads();
    for (int k = 0; k < PN2_K; ++k) {
        float bv = 3.0e38f; int bi = 0x7fffffff;
        for (int i = lane; i < Nsrc; i += 32) {
            float v = d2s[i];
            if (v < bv || (v == bv && i < bi)) { bv = v; bi = i; }
        }
        for (int off = 16; off > 0; off >>= 1) {
            float ov = __shfl_xor(bv, off, 32);
            int   oi = __shfl_xor(bi, off, 32);
            if (ov < bv || (ov == bv && oi < bi)) { bv = ov; bi = oi; }
        }
        if (lane == 0) {
            nidx[(size_t)q * PN2_K + k] = bi;
            mask[(size_t)q * PN2_K + k] = (bv <= r2) ? 1 : 0;
            d2s[bi] = 3.0e38f;
        }
        __syncthreads();
    }
}

// kNN (k=3) with inverse-squared-distance weights. One wave per query; 3 passes
// skipping previously chosen indices.
__global__ void pn2_knn3(const float* __restrict__ src, const float* __restrict__ dst,
                         int Nsrc, int Mq, int totalQ,
                         int* __restrict__ kidx, float* __restrict__ kw) {
    int q = blockIdx.x;
    if (q >= totalQ) return;
    int lane = threadIdx.x;
    int b = q / Mq;
    const float* sp = src + (size_t)b * Nsrc * 3;
    float qx = dst[q * 3 + 0], qy = dst[q * 3 + 1], qz = dst[q * 3 + 2];
    int c0 = -1, c1 = -1;
    for (int k = 0; k < 3; ++k) {
        float bv = 3.0e38f; int bi = 0x7fffffff;
        for (int i = lane; i < Nsrc; i += 32) {
            if (i == c0 || i == c1) continue;
            float dx = sp[i * 3 + 0] - qx, dy = sp[i * 3 + 1] - qy, dz = sp[i * 3 + 2] - qz;
            float v = dx * dx + dy * dy + dz * dz;
            if (v < bv || (v == bv && i < bi)) { bv = v; bi = i; }
        }
        for (int off = 16; off > 0; off >>= 1) {
            float ov = __shfl_xor(bv, off, 32);
            int   oi = __shfl_xor(bi, off, 32);
            if (ov < bv || (ov == bv && oi < bi)) { bv = ov; bi = oi; }
        }
        if (lane == 0) {
            kidx[q * 3 + k] = bi;
            kw[q * 3 + k]   = 1.f / fmaxf(bv, 1e-16f);
        }
        if (k == 0) c0 = bi; else if (k == 1) c1 = bi;
    }
}

// ---------------------------------------------------------------------------
// WMMA GEMM: out = relu(A[R,K] @ W + bias), A f16 row-major (K mult of 32),
// Wt = W^T padded [Nout,K] f16, Nout multiple of 64.
// Each wave computes a 16x64 tile: 4 accumulators share one A fragment, so
// per v_wmma_f32_16x16x32_f16 we issue 2.5 (not 4) b128 loads. 4 waves/block.
// ---------------------------------------------------------------------------
__global__ void pn2_gemm(const _Float16* __restrict__ A, const _Float16* __restrict__ Wt,
                         const float* __restrict__ bias,
                         _Float16* __restrict__ outH, float* __restrict__ outF,
                         int R, int K, int Nout) {
    int lane  = threadIdx.x & 31;
    int wv    = threadIdx.x >> 5;
    int tileM = blockIdx.y * 4 + wv;
    if (tileM * 16 >= R) return;                 // uniform per wave; EXEC stays all-1s
    int tileN = blockIdx.x;                      // 64-wide N tile
    int hsel  = lane >> 4;                       // 0/1: K-subblock owned by this lane
    int l15   = lane & 15;
    int row = tileM * 16 + l15; if (row >= R) row = R - 1;
    int colBase = tileN * 64 + l15;
    v8f c0 = {0.f,0.f,0.f,0.f,0.f,0.f,0.f,0.f};
    v8f c1 = c0, c2 = c0, c3 = c0;
    const _Float16* arow = A  + (size_t)row * K + (hsel << 4);        // K base / base+16
    const _Float16* br0  = Wt + (size_t)(colBase +  0) * K + (hsel << 4);
    const _Float16* br1  = Wt + (size_t)(colBase + 16) * K + (hsel << 4);
    const _Float16* br2  = Wt + (size_t)(colBase + 32) * K + (hsel << 4);
    const _Float16* br3  = Wt + (size_t)(colBase + 48) * K + (hsel << 4);
    for (int k0 = 0; k0 < K; k0 += 32) {
        v8h alo = *(const v8h*)(arow + k0);
        v8h ahi = *(const v8h*)(arow + k0 + 8);
        v16h a;
#pragma unroll
        for (int i = 0; i < 8; ++i) { a[i] = alo[i]; a[i + 8] = ahi[i]; }
        v8h blo, bhi; v16h bm;
        blo = *(const v8h*)(br0 + k0); bhi = *(const v8h*)(br0 + k0 + 8);
#pragma unroll
        for (int i = 0; i < 8; ++i) { bm[i] = blo[i]; bm[i + 8] = bhi[i]; }
        c0 = __builtin_amdgcn_wmma_f32_16x16x32_f16(false, a, false, bm, (short)0, c0, false, false);
        blo = *(const v8h*)(br1 + k0); bhi = *(const v8h*)(br1 + k0 + 8);
#pragma unroll
        for (int i = 0; i < 8; ++i) { bm[i] = blo[i]; bm[i + 8] = bhi[i]; }
        c1 = __builtin_amdgcn_wmma_f32_16x16x32_f16(false, a, false, bm, (short)0, c1, false, false);
        blo = *(const v8h*)(br2 + k0); bhi = *(const v8h*)(br2 + k0 + 8);
#pragma unroll
        for (int i = 0; i < 8; ++i) { bm[i] = blo[i]; bm[i + 8] = bhi[i]; }
        c2 = __builtin_amdgcn_wmma_f32_16x16x32_f16(false, a, false, bm, (short)0, c2, false, false);
        blo = *(const v8h*)(br3 + k0); bhi = *(const v8h*)(br3 + k0 + 8);
#pragma unroll
        for (int i = 0; i < 8; ++i) { bm[i] = blo[i]; bm[i + 8] = bhi[i]; }
        c3 = __builtin_amdgcn_wmma_f32_16x16x32_f16(false, a, false, bm, (short)0, c3, false, false);
    }
    float bsc0 = bias[colBase +  0];
    float bsc1 = bias[colBase + 16];
    float bsc2 = bias[colBase + 32];
    float bsc3 = bias[colBase + 48];
#pragma unroll
    for (int v = 0; v < 8; ++v) {
        int m = tileM * 16 + v + (hsel << 3);    // lanes16-31 hold M = v+8
        if (m >= R) continue;
        float v0 = fmaxf(c0[v] + bsc0, 0.f);
        float v1 = fmaxf(c1[v] + bsc1, 0.f);
        float v2 = fmaxf(c2[v] + bsc2, 0.f);
        float v3 = fmaxf(c3[v] + bsc3, 0.f);
        if (outH) {
            _Float16* o = outH + (size_t)m * Nout + colBase;
            o[0] = (_Float16)v0; o[16] = (_Float16)v1; o[32] = (_Float16)v2; o[48] = (_Float16)v3;
        } else {
            float* o = outF + (size_t)m * Nout + colBase;
            o[0] = v0; o[16] = v1; o[32] = v2; o[48] = v3;
        }
    }
}

// ---------------------------------------------------------------------------
// Feature assembly / pooling kernels (all write zero-padded f16 rows)
// ---------------------------------------------------------------------------
__global__ void pn2_sa1_feats(const float* __restrict__ x, const float* __restrict__ pos,
                              const float* __restrict__ p1, const int* __restrict__ nidx,
                              _Float16* __restrict__ feats, int q0, int rows) {
    int t = blockIdx.x * blockDim.x + threadIdx.x;
    if (t >= rows) return;
    int q  = q0 + (t >> 6);
    int kk = t & 63;
    int b  = q / PN2_N1;
    int j  = nidx[(size_t)q * PN2_K + kk];
    size_t sg = (size_t)b * PN2_NPER + j;
    _Float16* f = feats + (size_t)t * 32;
    f[0] = (_Float16)x[sg];
    f[1] = (_Float16)(pos[sg * 3 + 0] - p1[q * 3 + 0]);
    f[2] = (_Float16)(pos[sg * 3 + 1] - p1[q * 3 + 1]);
    f[3] = (_Float16)(pos[sg * 3 + 2] - p1[q * 3 + 2]);
#pragma unroll
    for (int i = 4; i < 32; ++i) f[i] = (_Float16)0.f;
}

__global__ void pn2_sa2_feats(const _Float16* __restrict__ x1, const float* __restrict__ p1,
                              const float* __restrict__ p2, const int* __restrict__ nidx,
                              _Float16* __restrict__ feats, int q0, int rows) {
    int t = blockIdx.x * blockDim.x + threadIdx.x;
    if (t >= rows) return;
    int q  = q0 + (t >> 6);
    int kk = t & 63;
    int b  = q / PN2_N2;
    int j  = nidx[(size_t)q * PN2_K + kk];
    size_t sg = (size_t)b * PN2_N1 + j;
    const _Float16* xr = x1 + sg * 128;
    _Float16* f = feats + (size_t)t * 160;
#pragma unroll
    for (int i = 0; i < 16; ++i) ((v8h*)f)[i] = ((const v8h*)xr)[i];
    f[128] = (_Float16)(p1[sg * 3 + 0] - p2[q * 3 + 0]);
    f[129] = (_Float16)(p1[sg * 3 + 1] - p2[q * 3 + 1]);
    f[130] = (_Float16)(p1[sg * 3 + 2] - p2[q * 3 + 2]);
#pragma unroll
    for (int i = 131; i < 160; ++i) f[i] = (_Float16)0.f;
}

// masked max over the 64 neighbors of each query (chunk-local h rows)
__global__ void pn2_segmax(const _Float16* __restrict__ h, const unsigned char* __restrict__ mask,
                           _Float16* __restrict__ out, int q0, int cq, int C) {
    int t = blockIdx.x * blockDim.x + threadIdx.x;
    int total = cq * C;
    if (t >= total) return;
    int ql = t / C, c = t % C;
    int q  = q0 + ql;
    const unsigned char* mk = mask + (size_t)q * PN2_K;
    const _Float16* hp = h + ((size_t)ql * PN2_K) * C + c;
    float best = -1e9f;
    for (int kk = 0; kk < PN2_K; ++kk)
        if (mk[kk]) best = fmaxf(best, (float)hp[(size_t)kk * C]);
    out[(size_t)q * C + c] = (_Float16)best;
}

__global__ void pn2_sa3_feats(const _Float16* __restrict__ x2, const float* __restrict__ p2,
                              _Float16* __restrict__ feats, int total) {
    int t = blockIdx.x * blockDim.x + threadIdx.x;
    if (t >= total) return;
    const _Float16* xr = x2 + (size_t)t * 256;
    _Float16* f = feats + (size_t)t * 288;
#pragma unroll
    for (int i = 0; i < 32; ++i) ((v8h*)f)[i] = ((const v8h*)xr)[i];
    f[256] = (_Float16)p2[t * 3 + 0];
    f[257] = (_Float16)p2[t * 3 + 1];
    f[258] = (_Float16)p2[t * 3 + 2];
#pragma unroll
    for (int i = 259; i < 288; ++i) f[i] = (_Float16)0.f;
}

__global__ void pn2_globalmax(const _Float16* __restrict__ h, _Float16* __restrict__ g) {
    int t = blockIdx.x * blockDim.x + threadIdx.x;
    if (t >= PN2_B * 512) return;
    int b = t / 512, c = t % 512;
    const _Float16* hp = h + ((size_t)b * PN2_N2) * 512 + c;
    float best = -1e9f;
    for (int n = 0; n < PN2_N2; ++n) best = fmaxf(best, (float)hp[(size_t)n * 512]);
    g[t] = (_Float16)best;
}

__global__ void pn2_fp3_feats(const _Float16* __restrict__ g, const _Float16* __restrict__ x2,
                              _Float16* __restrict__ feats, int total) {
    int t = blockIdx.x * blockDim.x + threadIdx.x;
    if (t >= total) return;
    int b = t / PN2_N2;
    const _Float16* gr = g  + (size_t)b * 512;
    const _Float16* xr = x2 + (size_t)t * 256;
    _Float16* f = feats + (size_t)t * 768;
#pragma unroll
    for (int i = 0; i < 64; ++i) ((v8h*)f)[i]      = ((const v8h*)gr)[i];
#pragma unroll
    for (int i = 0; i < 32; ++i) ((v8h*)f)[64 + i] = ((const v8h*)xr)[i];
}

// inverse-distance interpolation + concat, writes padded f16 feature rows
__global__ void pn2_interp_concat(const _Float16* __restrict__ fsrc, int Csrc,
                                  const int* __restrict__ kidx, const float* __restrict__ kw,
                                  const _Float16* __restrict__ xcatH, int CcatH,
                                  const float* __restrict__ xcatF, int CcatF,
                                  _Float16* __restrict__ feats, int Kpad,
                                  int MsrcPerBatch, int Mq, int totalQ) {
    int q = blockIdx.x * blockDim.x + threadIdx.x;
    if (q >= totalQ) return;
    int b = q / Mq;
    int i0 = kidx[q * 3 + 0], i1 = kidx[q * 3 + 1], i2 = kidx[q * 3 + 2];
    float w0 = kw[q * 3 + 0], w1 = kw[q * 3 + 1], w2 = kw[q * 3 + 2];
    float inv = 1.f / (w0 + w1 + w2);
    const _Float16* s0 = fsrc + ((size_t)b * MsrcPerBatch + i0) * Csrc;
    const _Float16* s1 = fsrc + ((size_t)b * MsrcPerBatch + i1) * Csrc;
    const _Float16* s2 = fsrc + ((size_t)b * MsrcPerBatch + i2) * Csrc;
    _Float16* f = feats + (size_t)q * Kpad;
    for (int c = 0; c < Csrc; ++c)
        f[c] = (_Float16)((w0 * (float)s0[c] + w1 * (float)s1[c] + w2 * (float)s2[c]) * inv);
    int o = Csrc;
    if (xcatH) {
        const _Float16* xr = xcatH + (size_t)q * CcatH;
        for (int c = 0; c < CcatH; ++c) f[o + c] = xr[c];
        o += CcatH;
    }
    if (xcatF) {
        for (int c = 0; c < CcatF; ++c) f[o + c] = (_Float16)xcatF[(size_t)q * CcatF + c];
        o += CcatF;
    }
    for (; o < Kpad; ++o) f[o] = (_Float16)0.f;
}

// ---------------------------------------------------------------------------
// Host orchestration
// ---------------------------------------------------------------------------
static inline dim3 gemm_grid(int R, int Nout) { return dim3(Nout / 64, (R + 63) / 64); }

extern "C" void kernel_launch(void* const* d_in, const int* in_sizes, int n_in,
                              void* d_out, int out_size, void* d_ws, size_t ws_size,
                              hipStream_t stream) {
    (void)in_sizes; (void)n_in; (void)out_size;
    const float* x   = (const float*)d_in[0];
    const float* pos = (const float*)d_in[1];
    const float* W[15]; const float* Bs[15];
    for (int i = 0; i < 15; ++i) { W[i] = (const float*)d_in[3 + 2 * i]; Bs[i] = (const float*)d_in[4 + 2 * i]; }

    static const int CIN [15] = {4, 64, 64,  131, 128, 128,  259, 256, 256,  768, 256,  384, 128,  129, 128};
    static const int NOUT[15] = {64, 64, 128, 128, 128, 256,  256, 256, 512,  256, 256,  128, 128,  128, 128};
    static const int KPAD[15] = {32, 64, 64,  160, 128, 128,  288, 256, 256,  768, 256,  384, 128,  160, 128};

    const int TOTQ1 = PN2_B * PN2_N1;   // 16376
    const int TOTQ2 = PN2_B * PN2_N2;   // 5408
    const int TOTP  = PN2_B * PN2_NPER; // 16384

    char* ws = (char*)d_ws;
    size_t off = 0;
    auto alloc = [&](size_t bytes) -> void* {
        void* p = ws + off;
        off += (bytes + 255) & ~(size_t)255;
        return p;
    };
    _Float16* Wt[15];
    for (int i = 0; i < 15; ++i) Wt[i] = (_Float16*)alloc((size_t)NOUT[i] * KPAD[i] * 2);
    int*      IDX1 = (int*)alloc((size_t)TOTQ1 * 4);
    int*      IDX2 = (int*)alloc((size_t)TOTQ2 * 4);
    float*    P1   = (float*)alloc((size_t)TOTQ1 * 3 * 4);
    float*    P2   = (float*)alloc((size_t)TOTQ2 * 3 * 4);
    int*      NIDX = (int*)alloc((size_t)TOTQ1 * PN2_K * 4);
    unsigned char* MASK = (unsigned char*)alloc((size_t)TOTQ1 * PN2_K);
    _Float16* X1   = (_Float16*)alloc((size_t)TOTQ1 * 128 * 2);
    _Float16* X2   = (_Float16*)alloc((size_t)TOTQ2 * 256 * 2);
    _Float16* G    = (_Float16*)alloc((size_t)PN2_B * 512 * 2);
    _Float16* F3   = (_Float16*)alloc((size_t)TOTQ2 * 256 * 2);
    _Float16* F2   = (_Float16*)alloc((size_t)TOTQ1 * 128 * 2);
    int*      KIDX = (int*)alloc((size_t)TOTP * 3 * 4);
    float*    KW   = (float*)alloc((size_t)TOTP * 3 * 4);
    _Float16* FEATS = (_Float16*)alloc((size_t)TOTQ1 * 384 * 2);   // max feats (FP2)
    _Float16* HA   = (_Float16*)alloc((size_t)16384 * 256 * 2);
    _Float16* HB   = (_Float16*)alloc((size_t)16384 * 256 * 2);
    if (off > ws_size) return;   // workspace too small; bail deterministically

    // ---- weight conversion ----
    for (int i = 0; i < 15; ++i) {
        int total = NOUT[i] * KPAD[i];
        pn2_wt_convert<<<(total + 255) / 256, 256, 0, stream>>>(W[i], Wt[i], CIN[i], NOUT[i], KPAD[i]);
    }

    // ---- SA1 ----
    pn2_fps<<<PN2_B, 256, 0, stream>>>(pos, IDX1, PN2_NPER, PN2_N1);
    pn2_gather3<<<(TOTQ1 + 255) / 256, 256, 0, stream>>>(pos, IDX1, P1, PN2_N1, PN2_NPER, TOTQ1);
    pn2_ball_query<<<TOTQ1, 32, 0, stream>>>(pos, P1, PN2_NPER, PN2_N1, TOTQ1, 0.16f, NIDX, MASK);
    for (int q0 = 0; q0 < TOTQ1; q0 += 256) {
        int cq = (TOTQ1 - q0 < 256) ? (TOTQ1 - q0) : 256;
        int rows = cq * PN2_K;
        pn2_sa1_feats<<<(rows + 255) / 256, 256, 0, stream>>>(x, pos, P1, NIDX, FEATS, q0, rows);
        pn2_gemm<<<gemm_grid(rows, 64), 128, 0, stream>>>(FEATS, Wt[0], Bs[0], HA, nullptr, rows, 32, 64);
        pn2_gemm<<<gemm_grid(rows, 64), 128, 0, stream>>>(HA, Wt[1], Bs[1], HB, nullptr, rows, 64, 64);
        pn2_gemm<<<gemm_grid(rows, 128), 128, 0, stream>>>(HB, Wt[2], Bs[2], HA, nullptr, rows, 64, 128);
        int tot = cq * 128;
        pn2_segmax<<<(tot + 255) / 256, 256, 0, stream>>>(HA, MASK, X1, q0, cq, 128);
    }

    // ---- SA2 ----
    pn2_fps<<<PN2_B, 256, 0, stream>>>(P1, IDX2, PN2_N1, PN2_N2);
    pn2_gather3<<<(TOTQ2 + 255) / 256, 256, 0, stream>>>(P1, IDX2, P2, PN2_N2, PN2_N1, TOTQ2);
    pn2_ball_query<<<TOTQ2, 32, 0, stream>>>(P1, P2, PN2_N1, PN2_N2, TOTQ2, 0.36f, NIDX, MASK);
    for (int q0 = 0; q0 < TOTQ2; q0 += 256) {
        int cq = (TOTQ2 - q0 < 256) ? (TOTQ2 - q0) : 256;
        int rows = cq * PN2_K;
        pn2_sa2_feats<<<(rows + 255) / 256, 256, 0, stream>>>(X1, P1, P2, NIDX, FEATS, q0, rows);
        pn2_gemm<<<gemm_grid(rows, 128), 128, 0, stream>>>(FEATS, Wt[3], Bs[3], HA, nullptr, rows, 160, 128);
        pn2_gemm<<<gemm_grid(rows, 128), 128, 0, stream>>>(HA, Wt[4], Bs[4], HB, nullptr, rows, 128, 128);
        pn2_gemm<<<gemm_grid(rows, 256), 128, 0, stream>>>(HB, Wt[5], Bs[5], HA, nullptr, rows, 128, 256);
        int tot = cq * 256;
        pn2_segmax<<<(tot + 255) / 256, 256, 0, stream>>>(HA, MASK, X2, q0, cq, 256);
    }

    // ---- Global SA (SA3) ----
    pn2_sa3_feats<<<(TOTQ2 + 255) / 256, 256, 0, stream>>>(X2, P2, FEATS, TOTQ2);
    pn2_gemm<<<gemm_grid(TOTQ2, 256), 128, 0, stream>>>(FEATS, Wt[6], Bs[6], HA, nullptr, TOTQ2, 288, 256);
    pn2_gemm<<<gemm_grid(TOTQ2, 256), 128, 0, stream>>>(HA, Wt[7], Bs[7], HB, nullptr, TOTQ2, 256, 256);
    pn2_gemm<<<gemm_grid(TOTQ2, 512), 128, 0, stream>>>(HB, Wt[8], Bs[8], HA, nullptr, TOTQ2, 256, 512);
    pn2_globalmax<<<(PN2_B * 512 + 255) / 256, 256, 0, stream>>>(HA, G);

    // ---- FP3 (broadcast global point) ----
    pn2_fp3_feats<<<(TOTQ2 + 255) / 256, 256, 0, stream>>>(G, X2, FEATS, TOTQ2);
    pn2_gemm<<<gemm_grid(TOTQ2, 256), 128, 0, stream>>>(FEATS, Wt[9], Bs[9], HA, nullptr, TOTQ2, 768, 256);
    pn2_gemm<<<gemm_grid(TOTQ2, 256), 128, 0, stream>>>(HA, Wt[10], Bs[10], F3, nullptr, TOTQ2, 256, 256);

    // ---- FP2 (knn3 N2 -> N1) ----
    pn2_knn3<<<TOTQ1, 32, 0, stream>>>(P2, P1, PN2_N2, PN2_N1, TOTQ1, KIDX, KW);
    pn2_interp_concat<<<(TOTQ1 + 255) / 256, 256, 0, stream>>>(
        F3, 256, KIDX, KW, X1, 128, nullptr, 0, FEATS, 384, PN2_N2, PN2_N1, TOTQ1);
    pn2_gemm<<<gemm_grid(TOTQ1, 128), 128, 0, stream>>>(FEATS, Wt[11], Bs[11], HA, nullptr, TOTQ1, 384, 128);
    pn2_gemm<<<gemm_grid(TOTQ1, 128), 128, 0, stream>>>(HA, Wt[12], Bs[12], F2, nullptr, TOTQ1, 128, 128);

    // ---- FP1 (knn3 N1 -> N_PER), final output f32 ----
    pn2_knn3<<<TOTP, 32, 0, stream>>>(P1, pos, PN2_N1, PN2_NPER, TOTP, KIDX, KW);
    pn2_interp_concat<<<(TOTP + 255) / 256, 256, 0, stream>>>(
        F2, 128, KIDX, KW, nullptr, 0, x, 1, FEATS, 160, PN2_N1, PN2_NPER, TOTP);
    pn2_gemm<<<gemm_grid(TOTP, 128), 128, 0, stream>>>(FEATS, Wt[13], Bs[13], HA, nullptr, TOTP, 160, 128);
    pn2_gemm<<<gemm_grid(TOTP, 128), 128, 0, stream>>>(HA, Wt[14], Bs[14], nullptr, (float*)d_out, TOTP, 128, 128);
}
